// MutualInformation_33509334843677
// MI455X (gfx1250) — matrix operand order repaired and checked by reading, not compiled, not run on hardware
//
#include <hip/hip_runtime.h>
#include <hip/hip_bf16.h>
#include <math.h>

typedef __attribute__((ext_vector_type(16))) _Float16 v16h;
typedef __attribute__((ext_vector_type(8)))  _Float16 v8h;
typedef __attribute__((ext_vector_type(8)))  float    v8f;

#define NUM_BINS   64
#define INV_SIGMA  10.0f          // 1/0.1
#define MI_EPS     1e-8f
#define NBATCH     2
#define CHUNK      32             // voxels per WMMA K-step (K=32 for f16 WMMA)
#define PAD        8              // row pad in halves -> 80B row stride (16B aligned)
#define WPB        4              // waves per block (main kernel)
#define MAIN_BLKX  96             // 96 blocks * 4 waves = 384 waves; 18432 chunks / 384 = 48 exact

// ---- workspace layout (32-bit words) ----
// [0 .. 8)              : u32 min/max keys, per batch {minF,maxF,minW,maxW}
// [WS_HF .. +B*64)      : histF (float)
// [WS_HW .. +B*64)      : histW (float)
// [WS_J  .. +B*64*64)   : joint (float)
#define WS_MM 0
#define WS_HF 8
#define WS_HW (WS_HF + NBATCH*NUM_BINS)
#define WS_J  (WS_HW + NBATCH*NUM_BINS)
#define WS_WORDS (WS_J + NBATCH*NUM_BINS*NUM_BINS)

union F16x16 { v16h v; v8h h[2]; };

// monotonic float<->uint key (order preserving) for deterministic integer min/max atomics
__device__ __forceinline__ unsigned f2key(float f) {
    unsigned u = __float_as_uint(f);
    return (u & 0x80000000u) ? ~u : (u | 0x80000000u);
}
__device__ __forceinline__ float key2f(unsigned k) {
    return (k & 0x80000000u) ? __uint_as_float(k & 0x7FFFFFFFu) : __uint_as_float(~k);
}

// ---------------- kernel 1: workspace init ----------------
__global__ void mi_init_kernel(unsigned* ws) {
    for (int i = blockIdx.x * blockDim.x + threadIdx.x; i < WS_WORDS;
         i += gridDim.x * blockDim.x) {
        if (i < 8) ws[i] = (i & 1) ? 0u : 0xFFFFFFFFu;   // even=min key, odd=max key
        else       ws[i] = 0u;                            // 0.0f bits
    }
}

// ---------------- kernel 2: per-batch, per-image min/max ----------------
__global__ void mi_minmax_kernel(const float* __restrict__ fixed,
                                 const float* __restrict__ warped,
                                 unsigned* __restrict__ ws, int N) {
    const int b   = blockIdx.y >> 1;
    const int img = blockIdx.y & 1;
    const float* __restrict__ src = (img ? warped : fixed) + (size_t)b * N;

    float lo = 3.4e38f, hi = -3.4e38f;
    for (int i = blockIdx.x * blockDim.x + threadIdx.x; i < N;
         i += gridDim.x * blockDim.x) {
        float v = src[i];
        lo = fminf(lo, v);
        hi = fmaxf(hi, v);
    }
    __shared__ float sLo[256], sHi[256];
    const int t = threadIdx.x;
    sLo[t] = lo; sHi[t] = hi;
    __syncthreads();
    for (int s = 128; s > 0; s >>= 1) {
        if (t < s) { sLo[t] = fminf(sLo[t], sLo[t + s]); sHi[t] = fmaxf(sHi[t], sHi[t + s]); }
        __syncthreads();
    }
    if (t == 0) {
        atomicMin(&ws[WS_MM + b * 4 + img * 2 + 0], f2key(sLo[0]));
        atomicMax(&ws[WS_MM + b * 4 + img * 2 + 1], f2key(sHi[0]));
    }
}

// ---------------- kernel 3: marginals + joint histogram via WMMA ----------------
// Per-wave private LDS staging: bin-major 64x32 f16 weight tiles (80B row stride,
// 16B aligned so all fragment/hist reads are ds_load_b128). The joint
// J = wf^T (64xN) * ww (Nx64) is tiled as 4x4 WMMA 16x16 tiles with K contracted
// 32 voxels at a time (v_wmma_f32_16x16x32_f16).
__global__ __launch_bounds__(WPB * 32)
void mi_main_kernel(const float* __restrict__ fixed,
                    const float* __restrict__ warped,
                    const unsigned* __restrict__ ws,
                    float* __restrict__ histF,
                    float* __restrict__ histW,
                    float* __restrict__ joint, int N) {
    alignas(16) __shared__ _Float16 sWF[WPB][NUM_BINS][CHUNK + PAD]; // [bin][voxel] f16
    alignas(16) __shared__ _Float16 sWW[WPB][NUM_BINS][CHUNK + PAD];
    __shared__ float sInvF[WPB][CHUNK];              // per-voxel 1/(sum+eps)
    __shared__ float sInvW[WPB][CHUNK];

    const int tid  = threadIdx.x;
    const int lane = tid & 31;
    const int wave = tid >> 5;
    const int b    = blockIdx.y;

    const float* __restrict__ fB = fixed  + (size_t)b * N;
    const float* __restrict__ wB = warped + (size_t)b * N;

    // uniform scalar loads of the per-batch min/max
    const float loF = key2f(ws[WS_MM + b * 4 + 0]);
    const float hiF = key2f(ws[WS_MM + b * 4 + 1]);
    const float loW = key2f(ws[WS_MM + b * 4 + 2]);
    const float hiW = key2f(ws[WS_MM + b * 4 + 3]);
    const float rnF = 1.0f / (hiF - loF + MI_EPS);
    const float rnW = 1.0f / (hiW - loW + MI_EPS);

    const int globalWave = blockIdx.x * WPB + wave;
    const int totalWaves = gridDim.x * WPB;
    const int numChunks  = (N + CHUNK - 1) / CHUNK;
    const int iters      = (numChunks + totalWaves - 1) / totalWaves; // uniform in block

    // 4x4 tiles of 16x16 f32 accumulators (128 VGPRs)
    v8f acc[4][4];
#pragma unroll
    for (int tm = 0; tm < 4; ++tm)
#pragma unroll
        for (int tn = 0; tn < 4; ++tn)
#pragma unroll
            for (int r = 0; r < 8; ++r) acc[tm][tn][r] = 0.0f;

    float hf0 = 0.f, hf1 = 0.f, hw0 = 0.f, hw1 = 0.f;   // marginal partials (2 bins/lane)

    for (int it = 0; it < iters; ++it) {
        const int  chunk  = globalWave + it * totalWaves;
        const bool active = chunk < numChunks;           // wave-uniform

        if (active) {
            const int  voxel = chunk * CHUNK + lane;
            const bool vok   = voxel < N;
            float xf = 0.f, xw = 0.f;
            if (vok) {
                xf = (fB[voxel] - loF) * rnF;
                xw = (wB[voxel] - loW) * rnW;
                // prefetch next chunk for this wave (global_prefetch_b8)
                if (voxel + CHUNK * totalWaves < N) {
                    __builtin_prefetch(&fB[voxel + CHUNK * totalWaves], 0, 3);
                    __builtin_prefetch(&wB[voxel + CHUNK * totalWaves], 0, 3);
                }
            }
            float sf = 0.f, sw = 0.f;
#pragma unroll 4
            for (int i = 0; i < NUM_BINS; ++i) {
                const float c  = (float)i * (1.0f / 63.0f);
                float df = (xf - c) * INV_SIGMA;
                float dw = (xw - c) * INV_SIGMA;
                float wfv = vok ? __expf(-0.5f * df * df) : 0.f;
                float wwv = vok ? __expf(-0.5f * dw * dw) : 0.f;
                sf += wfv; sw += wwv;
                sWF[wave][i][lane] = (_Float16)wfv;
                sWW[wave][i][lane] = (_Float16)wwv;
            }
            sInvF[wave][lane] = vok ? 1.0f / (sf + MI_EPS) : 0.f;
            sInvW[wave][lane] = vok ? 1.0f / (sw + MI_EPS) : 0.f;
        }
        __syncthreads();

        if (active) {
            // ---- marginals: lane owns bins i0, i0+1 ----
            // Full 32-half rows are contiguous & 16B aligned -> ds_load_b128.
            const int i0 = lane * 2;
            const v8h* rF0 = (const v8h*)&sWF[wave][i0][0];
            const v8h* rF1 = (const v8h*)&sWF[wave][i0 + 1][0];
            const v8h* rW0 = (const v8h*)&sWW[wave][i0][0];
            const v8h* rW1 = (const v8h*)&sWW[wave][i0 + 1][0];
#pragma unroll
            for (int blk = 0; blk < 4; ++blk) {
                v8h f0 = rF0[blk], f1 = rF1[blk];
                v8h w0 = rW0[blk], w1 = rW1[blk];
#pragma unroll
                for (int q = 0; q < 8; ++q) {
                    const int v  = blk * 8 + q;
                    const float isf = sInvF[wave][v];
                    const float isw = sInvW[wave][v];
                    hf0 += (float)f0[q] * isf;
                    hf1 += (float)f1[q] * isf;
                    hw0 += (float)w0[q] * isw;
                    hw1 += (float)w1[q] * isw;
                }
            }

            // ---- build fragments per ISA 16-bit A(16x32) / B(32x16) layouts ----
            // A lane (hi = lane>>4): halves {8*hi .. 8*hi+7} and {8*hi+16 .. 8*hi+23}
            //   -> v8h blocks hi and hi+2 of the bin row.
            // B lane: same k order as A -> v8h blocks hi and hi+2.
            const int rl = lane & 15;
            const int hi = lane >> 4;
            F16x16 aF[4], bF[4];
#pragma unroll
            for (int t = 0; t < 4; ++t) {
                const v8h* ra = (const v8h*)&sWF[wave][t * 16 + rl][0];
                const v8h* rb = (const v8h*)&sWW[wave][t * 16 + rl][0];
                aF[t].h[0] = ra[hi];
                aF[t].h[1] = ra[hi + 2];
                bF[t].h[0] = rb[hi];
                bF[t].h[1] = rb[hi + 2];
            }

            // ---- 16 WMMAs: acc[tm][tn] += A(tm) x B(tn) ----
#pragma unroll
            for (int tm = 0; tm < 4; ++tm)
#pragma unroll
                for (int tn = 0; tn < 4; ++tn)
                    acc[tm][tn] = __builtin_amdgcn_wmma_f32_16x16x32_f16(
                        false, aF[tm].v, false, bF[tn].v,
                        (short)0, acc[tm][tn], false, false);
        }
        __syncthreads();
    }

    // ---- flush partials ----
    float* __restrict__ jB = joint + (size_t)b * NUM_BINS * NUM_BINS;
    const int nn   = lane & 15;
    const int mOff = (lane >> 4) * 8;  // f32 C/D layout: lanes 16-31 hold M rows +8
#pragma unroll
    for (int tm = 0; tm < 4; ++tm)
#pragma unroll
        for (int tn = 0; tn < 4; ++tn)
#pragma unroll
            for (int r = 0; r < 8; ++r) {
                const int i = tm * 16 + mOff + r;
                const int j = tn * 16 + nn;
                atomicAdd(&jB[i * NUM_BINS + j], acc[tm][tn][r]);
            }
    const int i0 = lane * 2;
    atomicAdd(&histF[b * NUM_BINS + i0],     hf0);
    atomicAdd(&histF[b * NUM_BINS + i0 + 1], hf1);
    atomicAdd(&histW[b * NUM_BINS + i0],     hw0);
    atomicAdd(&histW[b * NUM_BINS + i0 + 1], hw1);
}

// ---------------- kernel 4: finalize MI ----------------
__device__ float blockSum(float v, float* sbuf) {
    const int t = threadIdx.x;
    sbuf[t] = v;
    __syncthreads();
    for (int s = 128; s > 0; s >>= 1) {
        if (t < s) sbuf[t] += sbuf[t + s];
        __syncthreads();
    }
    float r = sbuf[0];
    __syncthreads();
    return r;
}

__global__ void mi_finalize_kernel(const unsigned* __restrict__ ws, float* __restrict__ out) {
    const float* histF = (const float*)ws + WS_HF;
    const float* histW = (const float*)ws + WS_HW;
    const float* joint = (const float*)ws + WS_J;
    __shared__ float sbuf[256];
    const int t = threadIdx.x;

    float miSum = 0.f;
    for (int b = 0; b < NBATCH; ++b) {
        float s = 0.f;
        for (int i = t; i < NUM_BINS; i += blockDim.x) s += histF[b * NUM_BINS + i];
        const float sHF = blockSum(s, sbuf);
        s = 0.f;
        for (int i = t; i < NUM_BINS; i += blockDim.x) s += histW[b * NUM_BINS + i];
        const float sHW = blockSum(s, sbuf);
        s = 0.f;
        for (int i = t; i < NUM_BINS * NUM_BINS; i += blockDim.x)
            s += joint[b * NUM_BINS * NUM_BINS + i];
        const float sJ = blockSum(s, sbuf);

        const float invHF = 1.0f / sHF, invHW = 1.0f / sHW, invJ = 1.0f / sJ;
        float mi = 0.f;
        for (int idx = t; idx < NUM_BINS * NUM_BINS; idx += blockDim.x) {
            const int i = idx >> 6, j = idx & 63;
            const float pj = joint[b * NUM_BINS * NUM_BINS + idx] * invJ;
            const float pp = (histF[b * NUM_BINS + i] * invHF) *
                             (histW[b * NUM_BINS + j] * invHW);
            if (pj > MI_EPS && pp > MI_EPS) mi += pj * __logf(pj / pp);
        }
        miSum += blockSum(mi, sbuf);
    }
    if (t == 0) out[0] = -miSum / (float)NBATCH;
}

extern "C" void kernel_launch(void* const* d_in, const int* in_sizes, int n_in,
                              void* d_out, int out_size, void* d_ws, size_t ws_size,
                              hipStream_t stream) {
    const float* fixed  = (const float*)d_in[0];
    const float* warped = (const float*)d_in[1];
    float* out = (float*)d_out;

    const int total = in_sizes[0];
    const int N     = total / NBATCH;   // voxels per batch

    unsigned* ws = (unsigned*)d_ws;
    float* histF = (float*)ws + WS_HF;
    float* histW = (float*)ws + WS_HW;
    float* joint = (float*)ws + WS_J;

    mi_init_kernel<<<34, 256, 0, stream>>>(ws);
    mi_minmax_kernel<<<dim3(64, NBATCH * 2), 256, 0, stream>>>(fixed, warped, ws, N);
    mi_main_kernel<<<dim3(MAIN_BLKX, NBATCH), WPB * 32, 0, stream>>>(
        fixed, warped, ws, histF, histW, joint, N);
    mi_finalize_kernel<<<1, 256, 0, stream>>>(ws, out);
}
